// MultiHeadAttention_89086211653715
// MI455X (gfx1250) — compile-verified, matrix operand order checked by reading, not run on hardware
//
#include <hip/hip_runtime.h>
#include <hip/hip_bf16.h>

typedef __attribute__((ext_vector_type(16))) _Float16 v16h;
typedef __attribute__((ext_vector_type(8)))  _Float16 v8h;
typedef __attribute__((ext_vector_type(8)))  float    v8f;

#define EMBED 1024
#define HEADS 16
#define KVH   8
#define HD    64
#define SEQ   2048
#define NB    2

// ---------------------------------------------------------------------------
// LDS byte offset of a generic pointer known to point into LDS.
// ---------------------------------------------------------------------------
template <typename T>
__device__ __forceinline__ unsigned lds_off(T* p) {
  return (unsigned)(size_t)(__attribute__((address_space(3))) T*)p;
}

// Async global -> LDS copy, 16 bytes per lane (tracked by ASYNCcnt).
__device__ __forceinline__ void async_ld_b128(unsigned ldsoff, const void* g) {
  asm volatile("global_load_async_to_lds_b128 %0, %1, off"
               :
               : "v"(ldsoff), "v"(g)
               : "memory");
}
__device__ __forceinline__ void wait_async0() {
  asm volatile("s_wait_asynccnt 0" ::: "memory");
}

// ---------------------------------------------------------------------------
// Load a 16x32 (rows x K) f16 fragment in CDNA5 WMMA A/B VGPR layout.
// lane (l&15) = row; K chunks at (l>>4)*8 and (l>>4)*8+16 (two b128 loads).
// Works for global or LDS pointers (compiler infers ds_load for LDS).
// ---------------------------------------------------------------------------
__device__ __forceinline__ v16h load_frag32(const _Float16* __restrict__ p, int ld) {
  const int lane = threadIdx.x & 31;
  const _Float16* q = p + (size_t)(lane & 15) * (size_t)ld + ((lane >> 4) << 3);
  v16h f;
  *(v8h*)&f         = *(const v8h*)(q);
  *(((v8h*)&f) + 1) = *(const v8h*)(q + 16);
  return f;
}

__device__ __forceinline__ float row_max16(float v) {
  #pragma unroll
  for (int m = 1; m < 16; m <<= 1) v = fmaxf(v, __shfl_xor(v, m, 32));
  return v;
}
__device__ __forceinline__ float row_sum16(float v) {
  #pragma unroll
  for (int m = 1; m < 16; m <<= 1) v += __shfl_xor(v, m, 32);
  return v;
}

#define WMMA_F16(a, b, c) \
  __builtin_amdgcn_wmma_f32_16x16x32_f16(false, (a), false, (b), (short)0, (c), false, false)

// ---------------------------------------------------------------------------
// f32 -> f16 conversion
// ---------------------------------------------------------------------------
__global__ void cvt_f32_f16(const float* __restrict__ src, _Float16* __restrict__ dst, int n) {
  int i = blockIdx.x * blockDim.x + threadIdx.x;
  if (i < n) dst[i] = (_Float16)src[i];
}

// ---------------------------------------------------------------------------
// NT GEMM: C[m,n] = sum_k A[m,k] * B[n,k].
// Block = 8 waves -> 256(M) x 64(N) tile; wave = 32(M) x 64(N).
// Per 32-K step: B tile (64x32) staged in LDS via async b128 (double-
// buffered), 4 B frags preloaded to registers, 2 A frags register-pipelined
// across steps -> 8 WMMAs per wave per step.
// ---------------------------------------------------------------------------
__global__ void gemm_nt_wmma(const _Float16* __restrict__ A, int lda, long strideA,
                             const _Float16* __restrict__ B, int ldb,
                             _Float16* __restrict__ Ch, float* __restrict__ Cf,
                             const float* __restrict__ bias,
                             int ldm, int ldn, long strideC, int Kdim) {
  __shared__ __align__(16) _Float16 bstage[2][64 * 32];

  const int tid  = threadIdx.x;
  const int w    = tid >> 5;
  const int lane = tid & 31;
  const int m0 = blockIdx.x * 256 + w * 32;
  const int n0 = blockIdx.y * 64;
  const long aoff = (long)blockIdx.z * strideA + (long)m0 * lda;
  const long coff = (long)blockIdx.z * strideC;

  // cooperative staging: 64 rows x 32 halves = 4KB = 256 threads x 16B
  const int srow   = tid >> 2;
  const int schunk = (tid & 3) * 8;

  #define STAGE_B(buf, kk)                                                    \
    async_ld_b128(lds_off(&bstage[buf][srow * 32 + schunk]),                  \
                  B + (long)(n0 + srow) * ldb + (kk) + schunk)

  STAGE_B(0, 0);
  v16h a_cur0 = load_frag32(A + aoff, lda);
  v16h a_cur1 = load_frag32(A + aoff + (long)16 * lda, lda);
  wait_async0();
  __syncthreads();

  v8f acc[8] = {};
  const int nsteps = Kdim >> 5;
  for (int s = 0; s < nsteps; ++s) {
    v16h a_n0 = a_cur0, a_n1 = a_cur1;
    const bool more = (s + 1) < nsteps;
    if (more) {                       // issue next-step loads up front
      STAGE_B((s + 1) & 1, (s + 1) << 5);
      a_n0 = load_frag32(A + aoff + ((s + 1) << 5), lda);
      a_n1 = load_frag32(A + aoff + (long)16 * lda + ((s + 1) << 5), lda);
    }
    const _Float16* bb = &bstage[s & 1][0];
    v16h bfr[4];
    #pragma unroll
    for (int t = 0; t < 4; ++t) bfr[t] = load_frag32(bb + t * 16 * 32, 32);
    #pragma unroll
    for (int t = 0; t < 4; ++t) {
      acc[t]     = WMMA_F16(a_cur0, bfr[t], acc[t]);
      acc[4 + t] = WMMA_F16(a_cur1, bfr[t], acc[4 + t]);
    }
    if (more) wait_async0();
    __syncthreads();                  // all waves done with bstage[s&1]
    a_cur0 = a_n0; a_cur1 = a_n1;
  }
  #undef STAGE_B

  const int half_ = lane >> 4, nn = lane & 15;
  #pragma unroll
  for (int mh = 0; mh < 2; ++mh) {
    #pragma unroll
    for (int t = 0; t < 4; ++t) {
      #pragma unroll
      for (int r = 0; r < 8; ++r) {
        const int m = m0 + mh * 16 + r + 8 * half_;
        const int n = n0 + t * 16 + nn;
        const long ci = coff + (long)m * ldm + (long)n * ldn;
        const float v = acc[mh * 4 + t][r];
        if (Cf) Cf[ci] = v + (bias ? bias[n] : 0.0f);
        else    Ch[ci] = (_Float16)v;
      }
    }
  }
}

// ---------------------------------------------------------------------------
// Flash attention (non-causal symmetric ALiBi), GQA 16 heads / 8 kv heads.
// Block = 8 waves x 16 queries = 128 queries of one (batch, head).
// 64-key chunks: K (64x64) and V (64d x 64s, from transposed Vt) staged into
// LDS with async loads (4 b128/thread), double-buffered.  Per chunk per wave:
// 8 WMMAs energy + one online-softmax update over 64 cols + LDS repack of
// P (16x64) + 8 WMMAs P*V.  32 iterations, 32 barriers total.
// ---------------------------------------------------------------------------
__global__ void attn_fa_wmma(const _Float16* __restrict__ Q,
                             const _Float16* __restrict__ K,
                             const _Float16* __restrict__ Vt,
                             _Float16* __restrict__ O) {
  __shared__ __align__(16) _Float16 kstage[2][64 * 64];  // [key][d]
  __shared__ __align__(16) _Float16 vstage[2][64 * 64];  // [d][s-chunk]
  __shared__ __align__(16) _Float16 plds[8][16 * 64];

  const int tid   = threadIdx.x;
  const int w     = tid >> 5;
  const int lane  = tid & 31;
  const int half_ = lane >> 4, nn = lane & 15;
  const int h  = blockIdx.y;
  const int b  = blockIdx.z;
  const int q0 = blockIdx.x * 128 + w * 16;
  const int kvh = h >> 1;                          // repeat_interleave(kv, 2)

  const float slope = exp2f(-(float)(h + 1));      // 2^-(h+1), ALIBI_ALPHA=1
  const float CSC = 0.125f * 1.44269504088896340f; // (1/sqrt(64)) * log2(e)

  const long qbase = (long)b * SEQ * (HEADS * HD) + (long)q0 * (HEADS * HD) + h * HD;
  const long kbase = (long)b * SEQ * (KVH * HD) + kvh * HD;
  const long vbase = (long)b * (KVH * HD) * SEQ + (long)(kvh * HD) * SEQ;

  // staging map: 64 rows x 64 halves (8KB) = 512 b128 chunks; thread does
  // chunks tid and tid+256 -> rows cr and cr+32 at half-offset co.
  const int cr = tid >> 3, co = (tid & 7) * 8;

  #define STAGE_KV(buf, j)                                                      \
    do {                                                                        \
      async_ld_b128(lds_off(&kstage[buf][cr * 64 + co]),                        \
                    K + kbase + (long)((j) + cr) * (KVH * HD) + co);            \
      async_ld_b128(lds_off(&kstage[buf][(cr + 32) * 64 + co]),                 \
                    K + kbase + (long)((j) + cr + 32) * (KVH * HD) + co);       \
      async_ld_b128(lds_off(&vstage[buf][cr * 64 + co]),                        \
                    Vt + vbase + (long)cr * SEQ + (j) + co);                    \
      async_ld_b128(lds_off(&vstage[buf][(cr + 32) * 64 + co]),                 \
                    Vt + vbase + (long)(cr + 32) * SEQ + (j) + co);             \
    } while (0)

  const v16h a0 = load_frag32(Q + qbase,      HEADS * HD);  // d 0..31
  const v16h a1 = load_frag32(Q + qbase + 32, HEADS * HD);  // d 32..63

  STAGE_KV(0, 0);

  float mrow[8], lrow[8];
  v8f acc[4] = {};
  #pragma unroll
  for (int r = 0; r < 8; ++r) { mrow[r] = -1e30f; lrow[r] = 0.0f; }

  wait_async0();
  __syncthreads();

  for (int s = 0; s < (SEQ / 64); ++s) {
    const int j = s << 6;
    const bool more = (s + 1) < (SEQ / 64);
    if (more) STAGE_KV((s + 1) & 1, j + 64);   // overlap with this chunk

    const _Float16* kb = kstage[s & 1];
    const _Float16* vb = vstage[s & 1];

    // --- energy: four 16x16 tiles over key cols [j+16t, j+16t+16)
    v8f e[4];
    #pragma unroll
    for (int tp = 0; tp < 2; ++tp) {           // preload 4 frags, 4 WMMAs
      const int t0 = tp * 2, t1 = tp * 2 + 1;
      v16h klo0 = load_frag32(kb + (t0 * 16) * 64,      64);
      v16h khi0 = load_frag32(kb + (t0 * 16) * 64 + 32, 64);
      v16h klo1 = load_frag32(kb + (t1 * 16) * 64,      64);
      v16h khi1 = load_frag32(kb + (t1 * 16) * 64 + 32, 64);
      v8f z0 = {}, z1 = {};
      z0 = WMMA_F16(a0, klo0, z0);
      z1 = WMMA_F16(a0, klo1, z1);
      z0 = WMMA_F16(a1, khi0, z0);
      z1 = WMMA_F16(a1, khi1, z1);
      e[t0] = z0; e[t1] = z1;
    }

    // --- ALiBi bias + 1/8 scale folded into exp2 domain
    #pragma unroll
    for (int t = 0; t < 4; ++t) {
      #pragma unroll
      for (int r = 0; r < 8; ++r) {
        const float qi = (float)(q0 + r + 8 * half_);
        const float ki = (float)(j + t * 16 + nn);
        e[t][r] = (e[t][r] - slope * fabsf(qi - ki)) * CSC;
      }
    }

    // --- one online-softmax update per 64 keys
    float alpha[8];
    #pragma unroll
    for (int r = 0; r < 8; ++r) {
      float lm = fmaxf(fmaxf(e[0][r], e[1][r]), fmaxf(e[2][r], e[3][r]));
      const float mnew = fmaxf(mrow[r], row_max16(lm));
      const float a = exp2f(mrow[r] - mnew);
      e[0][r] = exp2f(e[0][r] - mnew);
      e[1][r] = exp2f(e[1][r] - mnew);
      e[2][r] = exp2f(e[2][r] - mnew);
      e[3][r] = exp2f(e[3][r] - mnew);
      lrow[r] = lrow[r] * a +
                row_sum16((e[0][r] + e[1][r]) + (e[2][r] + e[3][r]));
      mrow[r] = mnew;
      alpha[r] = a;
    }
    #pragma unroll
    for (int t = 0; t < 4; ++t)
      #pragma unroll
      for (int r = 0; r < 8; ++r) acc[t][r] *= alpha[r];

    // --- repack P (C-layout -> A-layout) through per-wave 16x64 LDS tile
    _Float16* pw = plds[w];
    #pragma unroll
    for (int t = 0; t < 4; ++t) {
      #pragma unroll
      for (int r = 0; r < 8; ++r) {
        pw[(r + 8 * half_) * 64 + t * 16 + nn] = (_Float16)e[t][r];
      }
    }
    const v16h pa0 = load_frag32(pw,      64);   // keys j..j+31
    const v16h pa1 = load_frag32(pw + 32, 64);   // keys j+32..j+63

    // --- acc += P (16x64) * V (64 keys x 64 d) from LDS
    #pragma unroll
    for (int t = 0; t < 4; ++t) {
      v16h blo = load_frag32(vb + (t * 16) * 64,      64);
      v16h bhi = load_frag32(vb + (t * 16) * 64 + 32, 64);
      acc[t] = WMMA_F16(pa0, blo, acc[t]);
      acc[t] = WMMA_F16(pa1, bhi, acc[t]);
    }

    if (more) wait_async0();
    __syncthreads();                      // all waves done with buffer s&1
  }
  #undef STAGE_KV

  // --- finalize: divide by row sums, store [b, q, h*64+d]
  #pragma unroll
  for (int t = 0; t < 4; ++t) {
    #pragma unroll
    for (int r = 0; r < 8; ++r) {
      const int q = q0 + r + 8 * half_;
      const int d = t * 16 + nn;
      O[(long)b * SEQ * (HEADS * HD) + (long)q * (HEADS * HD) + h * HD + d] =
          (_Float16)(acc[t][r] / lrow[r]);
    }
  }
}

// ---------------------------------------------------------------------------
// Launch: convert -> QKV projections -> flash attention -> output projection
// ---------------------------------------------------------------------------
extern "C" void kernel_launch(void* const* d_in, const int* in_sizes, int n_in,
                              void* d_out, int out_size, void* d_ws, size_t ws_size,
                              hipStream_t stream) {
  const float* x  = (const float*)d_in[0];
  const float* Wq = (const float*)d_in[1];
  const float* Wk = (const float*)d_in[2];
  const float* Wv = (const float*)d_in[3];
  const float* Wo = (const float*)d_in[4];
  const float* bo = (const float*)d_in[5];
  float* out = (float*)d_out;

  // f16 workspace layout (element offsets)
  _Float16* ws  = (_Float16*)d_ws;
  _Float16* xh  = ws;                 // N*S*E          = 4194304
  _Float16* Wqh = xh  + 4194304;      // 1024*1024      = 1048576
  _Float16* Wkh = Wqh + 1048576;      // 512*1024       =  524288
  _Float16* Wvh = Wkh + 524288;       // 512*1024       =  524288
  _Float16* Woh = Wvh + 524288;       // 1024*1024      = 1048576
  _Float16* Qh  = Woh + 1048576;      // [b,s,1024]     = 4194304
  _Float16* Kh  = Qh  + 4194304;      // [b,s,512]      = 2097152
  _Float16* Vth = Kh  + 2097152;      // [b,512,s] (Vt) = 2097152
  _Float16* AOh = Vth + 2097152;      // [b,s,1024]     = 4194304

  cvt_f32_f16<<<4194304 / 256, 256, 0, stream>>>(x,  xh,  4194304);
  cvt_f32_f16<<<1048576 / 256, 256, 0, stream>>>(Wq, Wqh, 1048576);
  cvt_f32_f16<<< 524288 / 256, 256, 0, stream>>>(Wk, Wkh,  524288);
  cvt_f32_f16<<< 524288 / 256, 256, 0, stream>>>(Wv, Wvh,  524288);
  cvt_f32_f16<<<1048576 / 256, 256, 0, stream>>>(Wo, Woh, 1048576);

  const dim3 blk(256, 1, 1);

  // Q = x @ Wq^T  -> [b, s, 1024]
  gemm_nt_wmma<<<dim3(SEQ / 256, 1024 / 64, NB), blk, 0, stream>>>(
      xh, EMBED, (long)SEQ * EMBED, Wqh, EMBED,
      Qh, nullptr, nullptr, 1024, 1, (long)SEQ * 1024, EMBED);
  // K = x @ Wk^T  -> [b, s, 512]
  gemm_nt_wmma<<<dim3(SEQ / 256, 512 / 64, NB), blk, 0, stream>>>(
      xh, EMBED, (long)SEQ * EMBED, Wkh, EMBED,
      Kh, nullptr, nullptr, 512, 1, (long)SEQ * 512, EMBED);
  // Vt = (x @ Wv^T)^T -> [b, 512, s]  (ldm=1 along s, ldn=SEQ along d)
  gemm_nt_wmma<<<dim3(SEQ / 256, 512 / 64, NB), blk, 0, stream>>>(
      xh, EMBED, (long)SEQ * EMBED, Wvh, EMBED,
      Vth, nullptr, nullptr, 1, SEQ, (long)512 * SEQ, EMBED);

  // flash attention with ALiBi
  attn_fa_wmma<<<dim3(SEQ / 128, HEADS, NB), blk, 0, stream>>>(Qh, Kh, Vth, AOh);

  // out = attn_out @ Wo^T + bo (f32), batches folded into M = 4096
  gemm_nt_wmma<<<dim3((NB * SEQ) / 256, 1024 / 64, 1), blk, 0, stream>>>(
      AOh, 1024, 0, Woh, 1024,
      nullptr, out, bo, 1024, 1, 0, 1024);
}
